// VectorQuantizer_81398220194537
// MI455X (gfx1250) — matrix operand order repaired and checked by reading, not compiled, not run on hardware
//
#include <hip/hip_runtime.h>
#include <hip/hip_bf16.h>

// ---------------------------------------------------------------------------
// VQ-VAE vector quantizer for MI455X (gfx1250, wave32).
//   z  : [B=8, D=256, T=2048] f32 ; emb: [n_e=8192, D=256] f32
// outputs (flat): z_q [B,D,T] f32, loss scalar, idx [B,T] (as f32)
//
// argmin_n ( ||e_n||^2 - 2 z.e_n ) via split-bf16 WMMA GEMM:
//   z.e ~= zhi.ehi + zhi.elo + zlo.ehi  (3 independent accumulators)
// A = emb tile (16 codes x K) from LDS; B = z (16 times x K) pinned in VGPRs.
// emb pre-packed to bf16 hi/lo tiles; hot loop stages tiles with the
// Tensor Data Mover (tensor_load_to_lds + s_wait_tensorcnt), double-buffered,
// with TDM pad fields inserting the 16B/row LDS bank-rotation padding.
// ---------------------------------------------------------------------------

typedef __attribute__((ext_vector_type(16))) __bf16 v16bf;
typedef __attribute__((ext_vector_type(8)))  float  v8f;
typedef __attribute__((ext_vector_type(4)))  unsigned int v4u;
typedef __attribute__((ext_vector_type(8)))  int    v8i;
typedef __attribute__((ext_vector_type(4)))  int    v4i;

#define B_  8
#define D_  256
#define T_  2048
#define NE  8192
#define BT  (B_ * T_)          // 16384
#define NEL (B_ * D_ * T_)     // 4194304
#define NTILE (NE / 16)        // 512 code tiles
#define TILE_BYTES (32 * D_ * 2)   // 16 KB packed hi+lo per tile

// ---------------------------------------------------------------- e2 = ||e||^2
__global__ void vq_e2_kernel(const float* __restrict__ emb, float* __restrict__ e2) {
  int n = blockIdx.x * blockDim.x + threadIdx.x;
  const float* row = emb + (size_t)n * D_;
  float s = 0.f;
  #pragma unroll 8
  for (int i = 0; i < D_; i += 4) {
    float4 v;
    __builtin_memcpy(&v, row + i, 16);
    s += v.x * v.x + v.y * v.y + v.z * v.z + v.w * v.w;
  }
  e2[n] = s;
}

// --------------------------------------------- pre-pack emb -> bf16 hi/lo tiles
// tile tt: hi rows [tt*8192 .. +4095], lo rows [tt*8192+4096 .. +8191]
__global__ void vq_pack_kernel(const float* __restrict__ emb, __bf16* __restrict__ embp) {
  const int tt = blockIdx.x;            // 0..NTILE-1
  const int sr = threadIdx.x >> 4;      // row 0..15
  const int sc = (threadIdx.x & 15) * 16;
  const float* src = emb + (size_t)(tt * 16 + sr) * D_ + sc;
  float4 xs[4];
  __builtin_memcpy(xs, src, 64);
  const float* xf = (const float*)xs;
  __bf16 hb[16], lb[16];
  #pragma unroll
  for (int j = 0; j < 16; ++j) {
    float x = xf[j];
    __bf16 h = (__bf16)x;
    hb[j] = h;
    lb[j] = (__bf16)(x - (float)h);
  }
  __bf16* dsth = embp + (size_t)tt * 8192 + sr * D_ + sc;
  __builtin_memcpy(dsth, hb, 32);
  __builtin_memcpy(dsth + 4096, lb, 32);
}

// ------------------------------------------------------------------ shared math
// Per-wave compute of one 16(code) x 16(time) tile + running argmin update.
// etile rows 0-15 = hi, 16-31 = lo, row stride 264 bf16 (528B, bank-rotating).
__device__ __forceinline__ void tile_compute(const __bf16 (*etile)[264],
                                             const v16bf* zh, const v16bf* zl,
                                             const float* e2, int n0, int mb,
                                             int ar, int hi16,
                                             float& bestd, int& besti) {
  v8f ahh = {0.f,0.f,0.f,0.f,0.f,0.f,0.f,0.f};
  v8f ahl = ahh, alh = ahh;
  #pragma unroll
  for (int kc = 0; kc < 8; ++kc) {
    const int cc = kc * 32 + (hi16 ? 8 : 0);
    union { v16bf v; char by[32]; } ah, al;
    __builtin_memcpy(&ah.by[0],  &etile[ar][cc],           16);
    __builtin_memcpy(&ah.by[16], &etile[ar][cc + 16],      16);
    __builtin_memcpy(&al.by[0],  &etile[16 + ar][cc],      16);
    __builtin_memcpy(&al.by[16], &etile[16 + ar][cc + 16], 16);
    ahh = __builtin_amdgcn_wmma_f32_16x16x32_bf16(false, ah.v, false, zh[kc],
                                                  (short)0, ahh, false, false);
    ahl = __builtin_amdgcn_wmma_f32_16x16x32_bf16(false, ah.v, false, zl[kc],
                                                  (short)0, ahl, false, false);
    alh = __builtin_amdgcn_wmma_f32_16x16x32_bf16(false, al.v, false, zh[kc],
                                                  (short)0, alh, false, false);
  }
  v8f acc = ahh + ahl + alh;
  float ev[8];
  __builtin_memcpy(&ev[0], e2 + n0 + mb,     16);
  __builtin_memcpy(&ev[4], e2 + n0 + mb + 4, 16);
  #pragma unroll
  for (int v = 0; v < 8; ++v) {
    float dist = fmaf(-2.0f, acc[v], ev[v]);
    int   code = n0 + mb + v;
    if (dist < bestd) { bestd = dist; besti = code; }
  }
}

// load this lane's persistent z operand (B matrix: 16 times x K=256, hi/lo)
__device__ __forceinline__ void load_z_regs(const float* __restrict__ z,
                                            int tglob, int hi16,
                                            v16bf* zh, v16bf* zl) {
  const int b    = tglob >> 11;
  const int tloc = tglob & (T_ - 1);
  const float* zb = z + ((size_t)b * D_) * T_ + tloc;
  #pragma unroll
  for (int kc = 0; kc < 8; ++kc) {
    const int kb = kc * 32 + (hi16 ? 8 : 0);
    #pragma unroll
    for (int j = 0; j < 16; ++j) {
      float x   = zb[(size_t)(kb + j + (j & 8)) * T_];
      __bf16 h  = (__bf16)x;
      zh[kc][j] = h;
      zl[kc][j] = (__bf16)(x - (float)h);
    }
  }
}

__device__ __forceinline__ void finish_argmin(float bestd, int besti,
                                              int lane, int w,
                                              int* idx_out, float* idxf_out) {
  float od = __shfl_xor(bestd, 16, 32);
  int   oi = __shfl_xor(besti, 16, 32);
  if (od < bestd || (od == bestd && oi < besti)) { bestd = od; besti = oi; }
  if (lane < 16) {
    int m = blockIdx.x * 128 + w * 16 + lane;
    idx_out[m]  = besti;
    idxf_out[m] = (float)besti;
  }
}

// ----------------------------------------------- TDM tensor-load tile descriptor
__device__ __forceinline__ void tdm_issue(unsigned long long gaddr, unsigned int lds_addr) {
  v4u g0;
  g0[0] = 1u;                                   // count=1 (valid user D#)
  g0[1] = lds_addr;                             // LDS byte address
  g0[2] = (unsigned int)gaddr;                  // global_addr[31:0] -> bits 95:64
  g0[3] = (unsigned int)((gaddr >> 32) & 0x01FFFFFFu) | 0x80000000u; // addr[56:32] | type=2
  // group1: data_size=2B(1), pad_enable, pad_interval=128dw(6), pad_amount=4dw(3)
  //         tensor_dim0=256 @bit48, tensor_dim1=32 @bit80,
  //         tile_dim0=256 @bit112, tile_dim1=32 @bit128, dim0_stride=256 @bit160
  v8i g1;
  g1[0] = 0x07910000; g1[1] = 0x01000000; g1[2] = 0x00200000; g1[3] = 0x01000000;
  g1[4] = 32;         g1[5] = 256;        g1[6] = 0;          g1[7] = 0;
  v4i z4 = {0, 0, 0, 0};
#if defined(__clang_major__) && (__clang_major__ >= 23)
  v8i z8 = {0, 0, 0, 0, 0, 0, 0, 0};
  __builtin_amdgcn_tensor_load_to_lds(g0, g1, z4, z4, z8, 0);
#else
  __builtin_amdgcn_tensor_load_to_lds(g0, g1, z4, z4, 0);
#endif
}

// ----------------------------------------------------- main pass (TDM staging)
__global__ __launch_bounds__(256)
void vq_argmin_tdm(const float* __restrict__ z,
                   const __bf16* __restrict__ embp,
                   const float* __restrict__ e2,
                   int*   __restrict__ idx_out,
                   float* __restrict__ idxf_out) {
  __shared__ __align__(16) __bf16 etile[2][32][264];   // double buffer, 33.8 KB

  const int tid  = threadIdx.x;
  const int lane = tid & 31;
  const int w    = tid >> 5;
  const int hi16 = (lane & 16) ? 1 : 0;
  const int ar   = lane & 15;
  const int mb   = hi16 ? 8 : 0;

  v16bf zh[8], zl[8];
  load_z_regs(z, blockIdx.x * 128 + w * 16 + ar, hi16, zh, zl);

  const unsigned long long gbase = (unsigned long long)(uintptr_t)embp;
  unsigned int lds0 = (unsigned int)(uintptr_t)&etile[0][0][0];
  unsigned int lds1 = (unsigned int)(uintptr_t)&etile[1][0][0];

  float bestd = 3.0e38f;
  int   besti = 0;

  if (w == 0) tdm_issue(gbase, lds0);                       // tile 0 -> buf 0
  for (int it = 0; it < NTILE; ++it) {
    const int cur = it & 1;
    __builtin_amdgcn_s_wait_tensorcnt(0);    // wave0: tile 'it' landed; others: nop
    __syncthreads();                         // all prev-tile readers done; data visible
    if (w == 0 && it + 1 < NTILE)
      tdm_issue(gbase + (unsigned long long)(it + 1) * TILE_BYTES,
                cur ? lds0 : lds1);          // overlap DMA of it+1 with compute
    tile_compute(etile[cur], zh, zl, e2, it * 16, mb, ar, hi16, bestd, besti);
  }
  finish_argmin(bestd, besti, lane, w, idx_out, idxf_out);
}

// --------------------------------------- fallback (self-staging, no packed ws)
__global__ __launch_bounds__(256)
void vq_argmin_fallback(const float* __restrict__ z,
                        const float* __restrict__ emb,
                        const float* __restrict__ e2,
                        int*   __restrict__ idx_out,
                        float* __restrict__ idxf_out) {
  __shared__ __align__(16) __bf16 etile[32][264];

  const int tid  = threadIdx.x;
  const int lane = tid & 31;
  const int w    = tid >> 5;
  const int hi16 = (lane & 16) ? 1 : 0;
  const int ar   = lane & 15;
  const int mb   = hi16 ? 8 : 0;

  v16bf zh[8], zl[8];
  load_z_regs(z, blockIdx.x * 128 + w * 16 + ar, hi16, zh, zl);

  float bestd = 3.0e38f;
  int   besti = 0;
  const int sr = tid >> 4;
  const int sc = (tid & 15) * 16;

  for (int n0 = 0; n0 < NE; n0 += 16) {
    const float* src = emb + (size_t)(n0 + sr) * D_ + sc;
    float4 xs[4];
    __builtin_memcpy(xs, src, 64);
    const float* xf = (const float*)xs;
    __bf16 hb[16], lb[16];
    #pragma unroll
    for (int j = 0; j < 16; ++j) {
      float x = xf[j];
      __bf16 h = (__bf16)x;
      hb[j] = h;
      lb[j] = (__bf16)(x - (float)h);
    }
    __builtin_memcpy(&etile[sr][sc], hb, 32);
    __builtin_memcpy(&etile[16 + sr][sc], lb, 32);
    if (n0 + 16 < NE) __builtin_prefetch(src + 16 * D_, 0, 1);
    __syncthreads();
    tile_compute(etile, zh, zl, e2, n0, mb, ar, hi16, bestd, besti);
    __syncthreads();
  }
  finish_argmin(bestd, besti, lane, w, idx_out, idxf_out);
}

// --------------------------------------- gather z_q = emb[idx], partial loss
__global__ void vq_gather_kernel(const float* __restrict__ z,
                                 const float* __restrict__ emb,
                                 const int*   __restrict__ idx,
                                 float* __restrict__ zq,
                                 float* __restrict__ partials) {
  const int tid = threadIdx.x;
  const int tq  = tid & 63;
  const int ds0 = tid >> 6;
  const int m   = blockIdx.x * 64 + tq;
  const int b   = m >> 11;
  const int t   = m & (T_ - 1);
  const int code = idx[m];

  const float* erow = emb + (size_t)code * D_;
  const float* zcol = z  + ((size_t)b * D_) * T_ + t;
  float*       ocol = zq + ((size_t)b * D_) * T_ + t;

  float err = 0.f;
  #pragma unroll 8
  for (int d = ds0; d < D_; d += 4) {
    float q = erow[d];
    float x = zcol[(size_t)d * T_];
    ocol[(size_t)d * T_] = q;
    float e = q - x;
    err += e * e;
  }
  __shared__ float red[256];
  red[tid] = err;
  __syncthreads();
  for (int s = 128; s > 0; s >>= 1) {
    if (tid < s) red[tid] += red[tid + s];
    __syncthreads();
  }
  if (tid == 0) partials[blockIdx.x] = red[0];
}

__global__ void vq_loss_kernel(const float* __restrict__ partials,
                               float* __restrict__ loss_out) {
  __shared__ float red[256];
  red[threadIdx.x] = partials[threadIdx.x];
  __syncthreads();
  for (int s = 128; s > 0; s >>= 1) {
    if (threadIdx.x < s) red[threadIdx.x] += red[threadIdx.x + s];
    __syncthreads();
  }
  if (threadIdx.x == 0) loss_out[0] = 1.25f * red[0] / (float)NEL;
}

// ---------------------------------------------------------------------------
extern "C" void kernel_launch(void* const* d_in, const int* in_sizes, int n_in,
                              void* d_out, int out_size, void* d_ws, size_t ws_size,
                              hipStream_t stream) {
  const float* z   = (const float*)d_in[0];
  const float* emb = (const float*)d_in[1];

  float* out  = (float*)d_out;
  float* zq   = out;
  float* loss = out + NEL;
  float* idxf = out + NEL + 1;

  char* ws = (char*)d_ws;
  float*  e2       = (float*)(ws);                 // 32 KB
  int*    idx      = (int*)  (ws + 32 * 1024);     // 64 KB
  float*  partials = (float*)(ws + 96 * 1024);     //  1 KB
  __bf16* embp     = (__bf16*)(ws + 128 * 1024);   //  8 MB packed tiles

  const size_t need_tdm = 128 * 1024 + (size_t)NTILE * TILE_BYTES;
  const bool use_tdm = ws_size >= need_tdm;

  vq_e2_kernel<<<NE / 256, 256, 0, stream>>>(emb, e2);
  if (use_tdm) {
    vq_pack_kernel<<<NTILE, 256, 0, stream>>>(emb, embp);
    vq_argmin_tdm<<<BT / 128, 256, 0, stream>>>(z, embp, e2, idx, idxf);
  } else {
    vq_argmin_fallback<<<BT / 128, 256, 0, stream>>>(z, emb, e2, idx, idxf);
  }
  vq_gather_kernel<<<BT / 64, 256, 0, stream>>>(z, emb, idx, zq, partials);
  vq_loss_kernel  <<<1,       256, 0, stream>>>(partials, loss);
}
